// AttnDecoderRNN_27556510171404
// MI455X (gfx1250) — matrix operand order repaired and checked by reading, not compiled, not run on hardware
//
#include <hip/hip_runtime.h>
#include <hip/hip_bf16.h>
#include <math.h>

// Problem dims (match reference)
#define Bd 32
#define Sd 128
#define Id 1024
#define Hd 1024
#define Ed 1024
#define Vd 32000

typedef __attribute__((ext_vector_type(16))) _Float16 v16h;
typedef __attribute__((ext_vector_type(8)))  float    v8f;

// ---------------------------------------------------------------------------
// A-tile (16x32 f16, MxK) lane layout used here (contiguous k order, same as B):
//   lanes 0-15 : M=lane,    elems i -> K=kb+i
//   lanes 16-31: M=lane-16, elems i -> K=kb+16+i
// ---------------------------------------------------------------------------
__device__ __forceinline__ v16h pack_a_tile(const float* __restrict__ A, int lda,
                                            int rowBase, int kb, int lane) {
    int m  = rowBase + (lane & 15);
    int k0 = kb + ((lane & 16) ? 16 : 0);
    const float4* p4 = reinterpret_cast<const float4*>(A + (long)m * lda + k0);
    float4 c0 = p4[0], c1 = p4[1];   // K = k0..k0+7
    float4 c2 = p4[2], c3 = p4[3];   // K = k0+8..k0+15
    v16h a;
    a[0]=(_Float16)c0.x; a[1]=(_Float16)c0.y; a[2]=(_Float16)c0.z; a[3]=(_Float16)c0.w;
    a[4]=(_Float16)c1.x; a[5]=(_Float16)c1.y; a[6]=(_Float16)c1.z; a[7]=(_Float16)c1.w;
    a[8]=(_Float16)c2.x; a[9]=(_Float16)c2.y; a[10]=(_Float16)c2.z; a[11]=(_Float16)c2.w;
    a[12]=(_Float16)c3.x; a[13]=(_Float16)c3.y; a[14]=(_Float16)c3.z; a[15]=(_Float16)c3.w;
    return a;
}

// B-tile (32x16 f16, KxN):
//   lanes 0-15 : N=lane,    elems i -> K=kb+i
//   lanes 16-31: N=lane-16, elems i -> K=kb+16+i
__device__ __forceinline__ v16h pack_b_contig(const float* __restrict__ p) {
    const float4* p4 = reinterpret_cast<const float4*>(p);
    float4 c0 = p4[0], c1 = p4[1], c2 = p4[2], c3 = p4[3];
    v16h b;
    b[0]=(_Float16)c0.x; b[1]=(_Float16)c0.y; b[2]=(_Float16)c0.z; b[3]=(_Float16)c0.w;
    b[4]=(_Float16)c1.x; b[5]=(_Float16)c1.y; b[6]=(_Float16)c1.z; b[7]=(_Float16)c1.w;
    b[8]=(_Float16)c2.x; b[9]=(_Float16)c2.y; b[10]=(_Float16)c2.z; b[11]=(_Float16)c2.w;
    b[12]=(_Float16)c3.x; b[13]=(_Float16)c3.y; b[14]=(_Float16)c3.z; b[15]=(_Float16)c3.w;
    return b;
}

// ---------------------------------------------------------------------------
// Pre-pack the [32,K] f32 activation into WMMA f16 lane layout:
//   packed[((kb/32)*2 + rowTile)*32 + lane] = v16h A-tile fragment
// One wave per (kChunk,rowTile); fully coalesced 32B/lane stores.
// ---------------------------------------------------------------------------
__global__ void pack_a_f16(const float* __restrict__ A, int lda, int K,
                           v16h* __restrict__ out) {
    int lane = threadIdx.x & 31;
    int gid  = blockIdx.x * (blockDim.x >> 5) + (threadIdx.x >> 5); // chunk*2+rt
    int nChunks = (K >> 5) * 2;
    if (gid >= nChunks) return;
    int chunk = gid >> 1;
    int rt    = gid & 1;
    out[(long)gid * 32 + lane] = pack_a_tile(A, lda, rt * 16, chunk * 32, lane);
}

// ---------------------------------------------------------------------------
// M=32 GEMM: out[32,N] = unpack(Ap) @ op(W) + bias
//   WMODE==0: W is [N, ldw] row-major, op(W)=W^T (16 contiguous f32/lane -> b128)
//   WMODE==1: W is [K, ldw] row-major, op(W)=W[:, coff:coff+N] (strided)
//   NT column tiles of 16 per wave; 2*NT v_wmma_f32_16x16x32_f16 per K-step.
// ---------------------------------------------------------------------------
template <int WMODE, int NT>
__global__ void gemm32_wmma(const v16h* __restrict__ Ap,
                            const float* __restrict__ W, int ldw, int coff,
                            const float* __restrict__ bias,
                            float* __restrict__ out, int ldo,
                            int N, int K) {
    int lane = threadIdx.x & 31;
    int wave = threadIdx.x >> 5;
    int tile = blockIdx.x * (blockDim.x >> 5) + wave;
    int col0 = tile * 16 * NT;
    if (col0 >= N) return;   // wave-uniform

    v8f acc[NT][2];
    #pragma unroll
    for (int t = 0; t < NT; ++t) { acc[t][0] = v8f{}; acc[t][1] = v8f{}; }

    int nLane = col0 + (lane & 15);
    int kSel  = (lane & 16) ? 16 : 0;

    for (int kb = 0, kc = 0; kb < K; kb += 32, ++kc) {
        v16h a0 = Ap[((long)kc * 2 + 0) * 32 + lane];
        v16h a1 = Ap[((long)kc * 2 + 1) * 32 + lane];
        #pragma unroll
        for (int t = 0; t < NT; ++t) {
            v16h bm;
            if (WMODE == 0) {
                bm = pack_b_contig(W + (long)(nLane + t * 16) * ldw + kb + kSel);
            } else {
                const float* p = W + (long)(kb + kSel) * ldw + coff + nLane + t * 16;
                #pragma unroll
                for (int i = 0; i < 16; ++i) bm[i] = (_Float16)p[(long)i * ldw];
            }
            acc[t][0] = __builtin_amdgcn_wmma_f32_16x16x32_f16(false, a0, false, bm,
                                                               (short)0, acc[t][0], false, false);
            acc[t][1] = __builtin_amdgcn_wmma_f32_16x16x32_f16(false, a1, false, bm,
                                                               (short)0, acc[t][1], false, false);
        }
    }

    // C/D layout: lanes 0-15 rows 0-7 (VGPR r -> M=r), lanes 16-31 rows 8-15.
    int mBase = (lane & 16) ? 8 : 0;
    #pragma unroll
    for (int t = 0; t < NT; ++t) {
        int n = nLane + t * 16;
        float bv = bias ? bias[n] : 0.0f;
        #pragma unroll
        for (int r = 0; r < 8; ++r) {
            out[(long)(mBase + r)      * ldo + n] = acc[t][0][r] + bv;
            out[(long)(16 + mBase + r) * ldo + n] = acc[t][1][r] + bv;
        }
    }
}

// ---- GRU elementwise (torch gate order r,z,n; biases fused here) ------------
__global__ void gru_elem(const float* __restrict__ xg, const float* __restrict__ hg,
                         const float* __restrict__ b_ih, const float* __restrict__ b_hh,
                         const float* __restrict__ h0, float* __restrict__ dec) {
    int idx = blockIdx.x * blockDim.x + threadIdx.x;
    if (idx >= Bd * Hd) return;
    int b = idx / Hd, h = idx % Hd;
    const float* xr = xg + (long)b * 3 * Hd;
    const float* hr = hg + (long)b * 3 * Hd;
    float rg = xr[h]      + b_ih[h]      + hr[h]      + b_hh[h];
    float zg = xr[Hd + h] + b_ih[Hd + h] + hr[Hd + h] + b_hh[Hd + h];
    float r  = 1.0f / (1.0f + __expf(-rg));
    float z  = 1.0f / (1.0f + __expf(-zg));
    float n  = tanhf(xr[2*Hd + h] + b_ih[2*Hd + h] + r * (hr[2*Hd + h] + b_hh[2*Hd + h]));
    dec[idx] = (1.0f - z) * n + z * h0[idx];
}

// ---- scores[b,s] = q_b . enc[s,b,:]  (one wave per (b,s) dot) ---------------
__global__ void scores_kernel(const float* __restrict__ q,
                              const float* __restrict__ enc,
                              float* __restrict__ scores) {
    int lane = threadIdx.x & 31;
    int wave = threadIdx.x >> 5;
    int idx  = blockIdx.x * (blockDim.x >> 5) + wave;
    if (idx >= Bd * Sd) return;
    int b = idx / Sd, s = idx % Sd;
    const float* qb = q + (long)b * Ed;
    const float* ep = enc + ((long)s * Bd + b) * Ed;
    float sum = 0.0f;
    for (int i = lane; i < Ed; i += 32) sum += qb[i] * ep[i];
    #pragma unroll
    for (int off = 16; off > 0; off >>= 1) sum += __shfl_down(sum, off, 32);
    if (lane == 0) scores[idx] = sum;
}

// ---- softmax over S per batch row; writes ws copy + tail of d_out -----------
__global__ void softmax_s(const float* __restrict__ scores,
                          float* __restrict__ attnw, float* __restrict__ out_attn) {
    __shared__ float red[Sd];
    int b = blockIdx.x, s = threadIdx.x;
    float v = scores[b * Sd + s];
    red[s] = v; __syncthreads();
    for (int off = Sd / 2; off > 0; off >>= 1) {
        if (s < off) red[s] = fmaxf(red[s], red[s + off]);
        __syncthreads();
    }
    float mx = red[0]; __syncthreads();
    float e = __expf(v - mx);
    red[s] = e; __syncthreads();
    for (int off = Sd / 2; off > 0; off >>= 1) {
        if (s < off) red[s] += red[s + off];
        __syncthreads();
    }
    float w = e / red[0];
    attnw[b * Sd + s]    = w;
    out_attn[b * Sd + s] = w;
}

// ---- context[b,e] = sum_s attnw[b,s] * enc[s,b,e] ---------------------------
__global__ void context_kernel(const float* __restrict__ attnw,
                               const float* __restrict__ enc,
                               float* __restrict__ ctx) {
    int idx = blockIdx.x * blockDim.x + threadIdx.x;   // b*E + e
    if (idx >= Bd * Ed) return;
    int b = idx / Ed, e = idx % Ed;
    float sum = 0.0f;
    for (int s = 0; s < Sd; ++s)
        sum += attnw[b * Sd + s] * enc[((long)s * Bd + b) * Ed + e];
    ctx[idx] = sum;
}

// ---- cat[b,:] = [x0[b,:], context[b,:]] -------------------------------------
__global__ void cat_kernel(const float* __restrict__ x0, const float* __restrict__ ctx,
                           float* __restrict__ cat) {
    int idx = blockIdx.x * blockDim.x + threadIdx.x;   // b*(I+E) + c
    if (idx >= Bd * (Id + Ed)) return;
    int b = idx / (Id + Ed), c = idx % (Id + Ed);
    cat[idx] = (c < Id) ? x0[(long)b * Id + c] : ctx[(long)b * Ed + (c - Id)];
}

// ---- log_softmax over V per batch row ---------------------------------------
__global__ void logsoftmax_v(const float* __restrict__ logits, float* __restrict__ out) {
    __shared__ float red[256];
    int b = blockIdx.x, tid = threadIdx.x;
    const float* row = logits + (long)b * Vd;
    float mx = -3.4e38f;
    for (int v = tid; v < Vd; v += 256) mx = fmaxf(mx, row[v]);
    red[tid] = mx; __syncthreads();
    for (int off = 128; off > 0; off >>= 1) {
        if (tid < off) red[tid] = fmaxf(red[tid], red[tid + off]);
        __syncthreads();
    }
    mx = red[0]; __syncthreads();
    float sum = 0.0f;
    for (int v = tid; v < Vd; v += 256) sum += __expf(row[v] - mx);
    red[tid] = sum; __syncthreads();
    for (int off = 128; off > 0; off >>= 1) {
        if (tid < off) red[tid] += red[tid + off];
        __syncthreads();
    }
    float lse = mx + logf(red[0]);
    for (int v = tid; v < Vd; v += 256)
        out[(long)b * Vd + v] = row[v] - lse;
}

// ---------------------------------------------------------------------------
extern "C" void kernel_launch(void* const* d_in, const int* in_sizes, int n_in,
                              void* d_out, int out_size, void* d_ws, size_t ws_size,
                              hipStream_t stream) {
    const float* x      = (const float*)d_in[0];   // [1,B,I]
    const float* enc    = (const float*)d_in[1];   // [S,B,E]
    const float* hidden = (const float*)d_in[2];   // [1,B,H]
    const float* W_ih   = (const float*)d_in[3];   // [3H, I]
    const float* W_hh   = (const float*)d_in[4];   // [3H, H]
    const float* b_ih   = (const float*)d_in[5];
    const float* b_hh   = (const float*)d_in[6];
    const float* W_attn = (const float*)d_in[7];   // [H, H+E]
    // d_in[8] = b_attn: cancels in softmax (constant per-row shift) -> unused
    const float* W_comb = (const float*)d_in[9];   // [I, I+E]
    const float* b_comb = (const float*)d_in[10];
    const float* W_out  = (const float*)d_in[11];  // [V, H]
    const float* b_out  = (const float*)d_in[12];

    float* out_logits = (float*)d_out;                  // [B,V]
    float* out_attn   = (float*)d_out + (long)Bd * Vd;  // [B,S]

    // workspace carve-up (floats; all offsets 32B-aligned)
    float* ws     = (float*)d_ws;
    float* xg     = ws;                       // 32*3072
    float* hg     = xg     + Bd * 3 * Hd;     // 32*3072
    float* dec    = hg     + Bd * 3 * Hd;     // 32*1024
    float* q      = dec    + Bd * Hd;         // 32*1024
    float* scores = q      + Bd * Ed;         // 32*128
    float* attnw  = scores + Bd * Sd;         // 32*128
    float* ctx    = attnw  + Bd * Sd;         // 32*1024
    float* cat    = ctx    + Bd * Ed;         // 32*2048
    float* comb   = cat    + Bd * (Id + Ed);  // 32*1024
    float* logits = comb   + Bd * Id;         // 32*32000
    // packed f16 A buffers (v16h = 32B units)
    v16h* packA   = (v16h*)(logits + (long)Bd * Vd);   // up to 32*2048 f16 = 128KB

    // 1) GRU gate GEMMs: pack activation, then [32,1024] @ [3072,1024]^T
    pack_a_f16<<<8, 256, 0, stream>>>(x, Id, Id, packA);
    gemm32_wmma<0, 2><<<12, 256, 0, stream>>>(packA, W_ih, Id, 0, nullptr, xg, 3 * Hd, 3 * Hd, Id);
    pack_a_f16<<<8, 256, 0, stream>>>(hidden, Hd, Hd, packA);
    gemm32_wmma<0, 2><<<12, 256, 0, stream>>>(packA, W_hh, Hd, 0, nullptr, hg, 3 * Hd, 3 * Hd, Hd);
    // 2) GRU elementwise -> dec
    gru_elem<<<(Bd * Hd) / 256, 256, 0, stream>>>(xg, hg, b_ih, b_hh, hidden, dec);
    // 3) q = dec @ W_attn[:, H:]  (non-transposed weight slice; strided B loads)
    pack_a_f16<<<8, 256, 0, stream>>>(dec, Hd, Hd, packA);
    gemm32_wmma<1, 1><<<8, 256, 0, stream>>>(packA, W_attn, Hd + Ed, Hd, nullptr, q, Ed, Ed, Hd);
    // 4) scores[b,s] = q_b . enc[s,b,:]
    scores_kernel<<<(Bd * Sd) / 8, 256, 0, stream>>>(q, enc, scores);
    // 5) softmax over S (also emits attn_weights output)
    softmax_s<<<Bd, Sd, 0, stream>>>(scores, attnw, out_attn);
    // 6) context = attnw @ enc
    context_kernel<<<(Bd * Ed) / 256, 256, 0, stream>>>(attnw, enc, ctx);
    // 7) cat = [x0, context]
    cat_kernel<<<(Bd * (Id + Ed)) / 256, 256, 0, stream>>>(x, ctx, cat);
    // 8) comb = cat @ W_comb^T + b_comb   (K=2048)
    pack_a_f16<<<16, 256, 0, stream>>>(cat, Id + Ed, Id + Ed, packA);
    gemm32_wmma<0, 2><<<4, 256, 0, stream>>>(packA, W_comb, Id + Ed, 0, b_comb, comb, Id, Id, Id + Ed);
    // 9) logits = comb @ W_out^T + b_out  (131 MB stream: the HBM-bound step)
    pack_a_f16<<<8, 256, 0, stream>>>(comb, Hd, Hd, packA);
    gemm32_wmma<0, 2><<<125, 256, 0, stream>>>(packA, W_out, Hd, 0, b_out, logits, Vd, Vd, Hd);
    // 10) log_softmax over V
    logsoftmax_v<<<Bd, 256, 0, stream>>>(logits, out_logits);
}
